// OpenAD_PN2_39539468927817
// MI455X (gfx1250) — matrix operand order repaired and checked by reading, not compiled, not run on hardware
//
#include <hip/hip_runtime.h>
#include <math.h>

// ---------------------------------------------------------------------------
// Types for CDNA5 WMMA
// ---------------------------------------------------------------------------
typedef __attribute__((ext_vector_type(16))) _Float16 v16h;
typedef __attribute__((ext_vector_type(8)))  float    v8f;

// ---------------------------------------------------------------------------
// CDNA5 async global->LDS staging (CDNA5 ISA / CDNA5 ISA).
// vdst = per-lane LDS byte address (low 32 bits of generic shared pointer,
// ISA 10.2: LDS aperture address keeps its low 32 bits); vaddr = 64-bit global.
// Tracked by ASYNCcnt; async loads complete in order within a wave, so with
// 2 issues per tile, "s_wait_asynccnt 2" waits for the previous tile while
// the just-issued tile stays in flight (software pipelining).
// ---------------------------------------------------------------------------
__device__ __forceinline__ void async_b128(void* lds, const void* g)
{
    asm volatile("global_load_async_to_lds_b128 %0, %1, off"
                 :: "v"((unsigned)(uintptr_t)lds), "v"(g)
                 : "memory");
}
__device__ __forceinline__ void wait_async(bool pipelined)
{
    if (pipelined) asm volatile("s_wait_asynccnt 2" ::: "memory");
    else           asm volatile("s_wait_asynccnt 0" ::: "memory");
}

// ---------------------------------------------------------------------------
// Fused GEMM (f16 x f16 -> f32) with optional bias, f16/f32 stores and
// per-channel sum / sum-of-squares accumulation for BatchNorm.
//   X  : [M, Kpad]  f16 row-major   (M multiple of 32, Kpad multiple of 32)
//   WT : [Npad, Kpad] f16           (row n = column n of W, zero padded)
//   Yh : [M, Npad] f16 (optional)   Yf : [M, Npad] f32 (optional)
//   stats: float[2*Npad] = {sum[c], sumsq[c]} (optional, must be pre-zeroed)
// Block = 256 threads = 8 waves in a 2(M) x 4(N) grid -> 32x64 block tile.
// X is staged once per K-step and shared by the 4 N-waves; LDS tiles are
// double-buffered with the async pipe one K-step ahead.
// grid = (M/32, ceil(Npad/64))
// ---------------------------------------------------------------------------
__global__ __launch_bounds__(256)
void gemm_wmma_kernel(const _Float16* __restrict__ X, const _Float16* __restrict__ WT,
                      const float* __restrict__ bias,
                      _Float16* __restrict__ Yh, float* __restrict__ Yf,
                      float* __restrict__ stats,
                      int M, int Kpad, int Npad)
{
    // 40-half row stride: 80B rows keep 16B alignment and spread bank starts.
    __shared__ _Float16 sX[2][32][40];
    __shared__ _Float16 sW[2][64][40];

    const int tid  = threadIdx.x;
    const int wave = tid >> 5;
    const int lane = tid & 31;
    const int mw   = wave & 1;    // M-wave: 0..1
    const int nw   = wave >> 1;   // N-wave: 0..3
    const int m0 = blockIdx.x * 32;
    const int n0 = blockIdx.y * 64;

    v8f acc = {};

    // fragment coordinates (CDNA5 16-bit A 16x32 / B 32x16 layouts)
    const int mrow = (mw << 4) + (lane & 15);
    const int kA   = (lane >> 4) << 3;        // 0 or 8
    const int brow = (nw << 4) + (lane & 15);
    const int bk   = (lane >> 4) << 4;        // 0 or 16

    // loader roles: waves 0-1 stage X (32x32), waves 2-5 stage WT (64x32)
    const bool loadX = (tid < 64);
    const int  xr = tid >> 1, xc = (tid & 1) * 16;
    const bool loadW = (tid >= 64 && tid < 192);
    const int  wi = tid - 64;
    const int  wn = wi >> 1, wc = (wi & 1) * 16;
    const bool wvalid = loadW && ((n0 + wn) < Npad);

    auto issue = [&](int buf, int k0) {
        if (loadX) {
            const _Float16* xs = X + (size_t)(m0 + xr) * Kpad + k0 + xc;
            async_b128(&sX[buf][xr][xc],     xs);
            async_b128(&sX[buf][xr][xc + 8], xs + 8);
            if (k0 + 64 < Kpad) __builtin_prefetch(xs + 64, 0, 1);
        }
        if (loadW) {
            if (wvalid) {
                const _Float16* wsrc = WT + (size_t)(n0 + wn) * Kpad + k0 + wc;
                async_b128(&sW[buf][wn][wc],     wsrc);
                async_b128(&sW[buf][wn][wc + 8], wsrc + 8);
            } else if (k0 < 32) {
                // N-edge pad rows: zero once, both buffers; never overwritten.
                const uint4 z = {0u, 0u, 0u, 0u};
                *(uint4*)&sW[0][wn][wc] = z; *(uint4*)&sW[0][wn][wc + 8] = z;
                *(uint4*)&sW[1][wn][wc] = z; *(uint4*)&sW[1][wn][wc + 8] = z;
            }
        }
    };

    issue(0, 0);
    int buf = 0;
    for (int k0 = 0; k0 < Kpad; k0 += 32) {
        const bool more = (k0 + 32) < Kpad;
        if (more) issue(buf ^ 1, k0 + 32);   // next tile in flight behind wmma
        wait_async(more);                    // previous tile has landed
        __syncthreads();                     // publish LDS tiles to all waves

        union { v16h v; uint4 q[2]; } a, b;
        // A: lane l holds row M=l%16, K = {kA..kA+7, kA+16..kA+23}
        a.q[0] = *(const uint4*)&sX[buf][mrow][kA];
        a.q[1] = *(const uint4*)&sX[buf][mrow][kA + 16];
        // B: lane l holds col N=l%16, K = {kA..kA+7, kA+16..kA+23}
        b.q[0] = *(const uint4*)&sW[buf][brow][kA];
        b.q[1] = *(const uint4*)&sW[buf][brow][kA + 16];

        acc = __builtin_amdgcn_wmma_f32_16x16x32_f16(
            /*neg_a=*/false, a.v, /*neg_b=*/false, b.v,
            /*c_mod=*/(short)0, acc, /*reuse_a=*/false, /*reuse_b=*/false);

        buf ^= 1;
        if (more) __syncthreads();           // guard buffer reissued next step
    }

    // epilogue: D layout — VGPR v, lane l -> M = v + 8*(l>>4), N = l%16
    const int col = n0 + (nw << 4) + (lane & 15);
    if (col >= Npad) return;
    const float bv = bias ? bias[col] : 0.0f;
    float psum = 0.f, psq = 0.f;
    const int mbase = m0 + (mw << 4) + ((lane >> 4) << 3);
    #pragma unroll
    for (int v = 0; v < 8; v++) {
        float y = acc[v] + bv;
        size_t o = (size_t)(mbase + v) * Npad + col;
        if (Yh) Yh[o] = (_Float16)y;
        if (Yf) Yf[o] = y;
        psum += y; psq += y * y;
    }
    if (stats) {
        atomicAdd(&stats[col], psum);
        atomicAdd(&stats[Npad + col], psq);
    }
}

// BatchNorm apply (+optional ReLU), in place on f16 activations.
// 8 halves (16B) per thread; Npad % 32 == 0 so a vector never crosses rows.
__global__ void bn_act_kernel(_Float16* __restrict__ Y, const float* __restrict__ stats,
                              const float* __restrict__ g, const float* __restrict__ beta,
                              float invCnt, int Npad, size_t total8, int relu)
{
    size_t i = (size_t)blockIdx.x * blockDim.x + threadIdx.x;
    if (i >= total8) return;
    size_t e0 = i * 8;
    int c0 = (int)(e0 % Npad);
    union { uint4 q; _Float16 h[8]; } u;
    u.q = *(uint4*)(Y + e0);
    #pragma unroll
    for (int j = 0; j < 8; j++) {
        int c = c0 + j;
        float m   = stats[c] * invCnt;
        float var = stats[Npad + c] * invCnt - m * m;
        float x = ((float)u.h[j] - m) * rsqrtf(var + 1e-5f) * g[c] + beta[c];
        if (relu) x = fmaxf(x, 0.f);
        u.h[j] = (_Float16)x;
    }
    *(uint4*)(Y + e0) = u.q;
}

// ---------------------------------------------------------------------------
// Weight / vector prep: W[K,N] f32 -> WT[Npad][Kpad] f16 (zero padded)
// ---------------------------------------------------------------------------
__global__ void prep_w_kernel(const float* __restrict__ w, int K, int N,
                              int Kpad, int Npad, _Float16* __restrict__ WT)
{
    int t = blockIdx.x * blockDim.x + threadIdx.x;
    if (t >= Kpad * Npad) return;
    int n = t / Kpad, k = t % Kpad;
    float v = (k < K && n < N) ? w[(size_t)k * N + n] : 0.f;
    WT[(size_t)n * Kpad + k] = (_Float16)v;
}

__global__ void prep_vec_kernel(const float* __restrict__ s, int N, int Npad,
                                float* __restrict__ d)
{
    int t = blockIdx.x * blockDim.x + threadIdx.x;
    if (t >= Npad) return;
    d[t] = (t < N) ? s[t] : 0.f;
}

// ---------------------------------------------------------------------------
// Point-cloud utility kernels
// ---------------------------------------------------------------------------
__global__ void transpose_xyz_kernel(const float* __restrict__ xyz, float* __restrict__ pts)
{
    int t = blockIdx.x * blockDim.x + threadIdx.x;           // B*N*3
    if (t >= 32 * 2048 * 3) return;
    int d = t % 3; int n = (t / 3) % 2048; int b = t / (3 * 2048);
    pts[t] = xyz[((size_t)b * 3 + d) * 2048 + n];
}

// farthest point sampling: one block per batch, dist kept in registers
__global__ __launch_bounds__(256)
void fps_kernel(const float* __restrict__ pts, int N, int S, int* __restrict__ idx)
{
    const int b = blockIdx.x;
    const float* p = pts + (size_t)b * N * 3;
    __shared__ float sred[256];
    __shared__ int   sidx[256];
    __shared__ float lastPt[3];
    float dist[8];
    const int per = N / 256;
    for (int j = 0; j < per; j++) dist[j] = 1e10f;
    int last = 0;
    if (threadIdx.x == 0) idx[b * S] = 0;
    for (int s = 1; s < S; s++) {
        if (threadIdx.x < 3) lastPt[threadIdx.x] = p[last * 3 + threadIdx.x];
        __syncthreads();
        float best = -1.f; int bi = 0;
        for (int j = 0; j < per; j++) {
            int n = threadIdx.x * per + j;
            float dx = p[n * 3] - lastPt[0];
            float dy = p[n * 3 + 1] - lastPt[1];
            float dz = p[n * 3 + 2] - lastPt[2];
            float d = dx * dx + dy * dy + dz * dz;
            dist[j] = fminf(dist[j], d);
            if (dist[j] > best) { best = dist[j]; bi = n; }
        }
        sred[threadIdx.x] = best; sidx[threadIdx.x] = bi;
        __syncthreads();
        for (int off = 128; off > 0; off >>= 1) {
            if (threadIdx.x < off) {
                float o = sred[threadIdx.x + off];
                int   oi = sidx[threadIdx.x + off];
                if (o > sred[threadIdx.x] ||
                    (o == sred[threadIdx.x] && oi < sidx[threadIdx.x])) {
                    sred[threadIdx.x] = o; sidx[threadIdx.x] = oi;
                }
            }
            __syncthreads();
        }
        last = sidx[0];
        if (threadIdx.x == 0) idx[b * S + s] = last;
        __syncthreads();
    }
}

__global__ void gather_xyz_kernel(const float* __restrict__ pts, const int* __restrict__ idx,
                                  int Nsrc, int S, float* __restrict__ out, int total)
{
    int t = blockIdx.x * blockDim.x + threadIdx.x;
    if (t >= total) return;
    int b = t / S;
    int n = idx[t];
    const float* p = pts + ((size_t)b * Nsrc + n) * 3;
    out[t * 3 + 0] = p[0]; out[t * 3 + 1] = p[1]; out[t * 3 + 2] = p[2];
}

// first-nsample-in-range (ascending index) ball query, padded with first hit
__global__ void ball_query_kernel(const float* __restrict__ xyz, const float* __restrict__ nxyz,
                                  int N, int S, int K, float r2, int* __restrict__ gi, int total)
{
    int t = blockIdx.x * blockDim.x + threadIdx.x;
    if (t >= total) return;
    int b = t / S;
    const float* p = xyz + (size_t)b * N * 3;
    float cx = nxyz[(size_t)t * 3], cy = nxyz[(size_t)t * 3 + 1], cz = nxyz[(size_t)t * 3 + 2];
    int* out = gi + (size_t)t * K;
    int cnt = 0, first = 0;
    for (int n = 0; n < N && cnt < K; n++) {
        float dx = p[n * 3] - cx, dy = p[n * 3 + 1] - cy, dz = p[n * 3 + 2] - cz;
        float d = dx * dx + dy * dy + dz * dz;
        if (d <= r2) { if (cnt == 0) first = n; out[cnt++] = n; }
    }
    for (; cnt < K; cnt++) out[cnt] = first;
}

// build grouped input rows: [feat(Cf) | xyz_rel(3) | zero pad] -> f16
__global__ void group_gather_kernel(const float* __restrict__ feat, int Nsrc, int Cf,
                                    const float* __restrict__ xyz, const float* __restrict__ nxyz,
                                    const int* __restrict__ gi, int S, int K, int Cpad,
                                    _Float16* __restrict__ X, size_t totalRows)
{
    size_t row = (size_t)blockIdx.x * blockDim.x + threadIdx.x;
    if (row >= totalRows) return;
    size_t bs = row / K;
    int s = (int)(bs % S); int b = (int)(bs / S);
    int n = gi[row];
    const float* f = feat + ((size_t)b * Nsrc + n) * Cf;
    _Float16* xr = X + row * Cpad;
    for (int c = 0; c < Cf; c++) xr[c] = (_Float16)f[c];
    const float* pp = xyz + ((size_t)b * Nsrc + n) * 3;
    const float* q  = nxyz + ((size_t)b * S + s) * 3;
    xr[Cf + 0] = (_Float16)(pp[0] - q[0]);
    xr[Cf + 1] = (_Float16)(pp[1] - q[1]);
    xr[Cf + 2] = (_Float16)(pp[2] - q[2]);
    for (int c = Cf + 3; c < Cpad; c++) xr[c] = (_Float16)0.f;
}

// max over nsample (post-ReLU f16 acts) -> f32 feature slice
__global__ void maxpool_kernel(const _Float16* __restrict__ Y, int K, int C, int Npad,
                               float* __restrict__ out, int outStride, int colOff, size_t BS)
{
    size_t t = (size_t)blockIdx.x * blockDim.x + threadIdx.x;
    if (t >= BS * C) return;
    size_t bs = t / C; int c = (int)(t % C);
    const _Float16* y = Y + (bs * K) * (size_t)Npad + c;
    float m = -3.4e38f;
    for (int k = 0; k < K; k++) m = fmaxf(m, (float)y[(size_t)k * Npad]);
    out[bs * outStride + colOff + c] = m;
}

// sa3 group-all rows: [xyz(3) | feat(512) | pad..544]
__global__ void sa3_build_kernel(const float* __restrict__ l2x, const float* __restrict__ l2f,
                                 _Float16* __restrict__ X)
{
    int row = blockIdx.x * blockDim.x + threadIdx.x;
    if (row >= 32 * 128) return;
    _Float16* xr = X + (size_t)row * 544;
    const float* p = l2x + (size_t)row * 3;
    xr[0] = (_Float16)p[0]; xr[1] = (_Float16)p[1]; xr[2] = (_Float16)p[2];
    const float* f = l2f + (size_t)row * 512;
    for (int c = 0; c < 512; c++) xr[3 + c] = (_Float16)f[c];
    for (int c = 515; c < 544; c++) xr[c] = (_Float16)0.f;
}

// fp3 rows: [l2f(512) | broadcast l3f(1024)]
__global__ void fp3_build_kernel(const float* __restrict__ l2f, const float* __restrict__ l3f,
                                 _Float16* __restrict__ X)
{
    int row = blockIdx.x * blockDim.x + threadIdx.x;
    if (row >= 32 * 128) return;
    int b = row >> 7;
    _Float16* xr = X + (size_t)row * 1536;
    const float* f = l2f + (size_t)row * 512;
    for (int c = 0; c < 512; c++) xr[c] = (_Float16)f[c];
    const float* g = l3f + (size_t)b * 1024;
    for (int c = 0; c < 1024; c++) xr[512 + c] = (_Float16)g[c];
}

// copy f32 features into leading columns of an f16 row
__global__ void copy_cols_kernel(const float* __restrict__ src, int C, int srcStride,
                                 _Float16* __restrict__ X, int Cpad, size_t rows)
{
    size_t row = (size_t)blockIdx.x * blockDim.x + threadIdx.x;
    if (row >= rows) return;
    const float* s = src + row * (size_t)srcStride;
    _Float16* xr = X + row * (size_t)Cpad;
    for (int c = 0; c < C; c++) xr[c] = (_Float16)s[c];
}

// fp1 rows: [pts(3) | pts(3) | (interp fills 6..134) | pad 134..160]
__global__ void fp1_build_kernel(const float* __restrict__ pts, _Float16* __restrict__ X)
{
    int row = blockIdx.x * blockDim.x + threadIdx.x;
    if (row >= 32 * 2048) return;
    const float* p = pts + (size_t)row * 3;
    _Float16* xr = X + (size_t)row * 160;
    xr[0] = (_Float16)p[0]; xr[1] = (_Float16)p[1]; xr[2] = (_Float16)p[2];
    xr[3] = (_Float16)p[0]; xr[4] = (_Float16)p[1]; xr[5] = (_Float16)p[2];
    for (int c = 134; c < 160; c++) xr[c] = (_Float16)0.f;
}

// 3-NN inverse-squared-distance interpolation, written into X columns
__global__ void knn3_interp_kernel(const float* __restrict__ xyz1, int N1,
                                   const float* __restrict__ xyz2, int N2,
                                   const _Float16* __restrict__ feat2, int C2, int fstride,
                                   _Float16* __restrict__ X, int Cpad, int colOff, int total)
{
    int t = blockIdx.x * blockDim.x + threadIdx.x;
    if (t >= total) return;
    int b = t / N1;
    const float* q = xyz1 + (size_t)t * 3;
    float qx = q[0], qy = q[1], qz = q[2];
    const float* p = xyz2 + (size_t)b * N2 * 3;
    float d0 = 3.4e38f, d1 = 3.4e38f, d2 = 3.4e38f;
    int i0 = 0, i1 = 0, i2 = 0;
    for (int j = 0; j < N2; j++) {
        float dx = p[j * 3] - qx, dy = p[j * 3 + 1] - qy, dz = p[j * 3 + 2] - qz;
        float d = dx * dx + dy * dy + dz * dz;
        if (d < d0)      { d2 = d1; i2 = i1; d1 = d0; i1 = i0; d0 = d; i0 = j; }
        else if (d < d1) { d2 = d1; i2 = i1; d1 = d; i1 = j; }
        else if (d < d2) { d2 = d; i2 = j; }
    }
    float w0 = 1.f / (fmaxf(d0, 0.f) + 1e-8f);
    float w1 = 1.f / (fmaxf(d1, 0.f) + 1e-8f);
    float w2 = 1.f / (fmaxf(d2, 0.f) + 1e-8f);
    float ws = w0 + w1 + w2; w0 /= ws; w1 /= ws; w2 /= ws;
    const _Float16* f0 = feat2 + ((size_t)b * N2 + i0) * fstride;
    const _Float16* f1 = feat2 + ((size_t)b * N2 + i1) * fstride;
    const _Float16* f2 = feat2 + ((size_t)b * N2 + i2) * fstride;
    _Float16* xr = X + (size_t)t * Cpad + colOff;
    for (int c = 0; c < C2; c++)
        xr[c] = (_Float16)(w0 * (float)f0[c] + w1 * (float)f1[c] + w2 * (float)f2[c]);
}

__global__ void rownorm_kernel(const _Float16* __restrict__ h, int C,
                               float* __restrict__ hn, int rows)
{
    int t = blockIdx.x * blockDim.x + threadIdx.x;
    if (t >= rows) return;
    const _Float16* r = h + (size_t)t * C;
    float s = 0.f;
    for (int c = 0; c < C; c++) { float v = (float)r[c]; s += v * v; }
    hn[t] = sqrtf(s);
}

__global__ void tn_kernel(const float* __restrict__ text, float* __restrict__ tn)
{
    int k = threadIdx.x;
    if (k >= 19) return;
    float s = 0.f;
    for (int d = 0; d < 512; d++) { float v = text[d * 19 + k]; s += v * v; }
    tn[k] = sqrtf(s);
}

// cosine logits + log_softmax over the 19 classes, transposed output [B,19,N]
__global__ void logits_kernel(const float* __restrict__ sim, const float* __restrict__ hn,
                              const float* __restrict__ tn, const float* __restrict__ ls,
                              float* __restrict__ out)
{
    int t = blockIdx.x * blockDim.x + threadIdx.x;
    if (t >= 32 * 2048) return;
    int b = t >> 11, n = t & 2047;
    float scale = ls[0] / hn[t];
    float x[19]; float mx = -3.4e38f;
    for (int k = 0; k < 19; k++) {
        float v = sim[(size_t)t * 32 + k] * scale / tn[k];
        x[k] = v; if (v > mx) mx = v;
    }
    float se = 0.f;
    for (int k = 0; k < 19; k++) se += expf(x[k] - mx);
    float lse = mx + logf(se);
    for (int k = 0; k < 19; k++)
        out[((size_t)b * 19 + k) * 2048 + n] = x[k] - lse;
}

// ---------------------------------------------------------------------------
// Host orchestration
// ---------------------------------------------------------------------------
extern "C" void kernel_launch(void* const* d_in, const int* in_sizes, int n_in,
                              void* d_out, int out_size, void* d_ws, size_t ws_size,
                              hipStream_t stream)
{
    (void)in_sizes; (void)n_in; (void)out_size; (void)ws_size;
    const float* xyz  = (const float*)d_in[0];
    const float* text = (const float*)d_in[1];
    auto P = [&](int i) { return (const float*)d_in[i]; };

    char* base = (char*)d_ws;
    size_t cur = 0;
    auto alloc = [&](size_t bytes) -> void* {
        void* p = base + cur;
        cur = (cur + bytes + 255) & ~(size_t)255;
        return p;
    };

    // f16 ping-pong activation buffers (each holds 2M x 128 halves = 512MB)
    const size_t ACT_HALF = (size_t)32 * 512 * 128 * 128;
    _Float16* ACT0 = (_Float16*)alloc(ACT_HALF * 2);
    _Float16* ACT1 = (_Float16*)alloc(ACT_HALF * 2);

    float* pts = (float*)alloc((size_t)32 * 2048 * 3 * 4);
    float* l1x = (float*)alloc((size_t)32 * 512 * 3 * 4);
    float* l2x = (float*)alloc((size_t)32 * 128 * 3 * 4);
    float* l1f = (float*)alloc((size_t)32 * 512 * 320 * 4);
    float* l2f = (float*)alloc((size_t)32 * 128 * 512 * 4);
    float* l3f = (float*)alloc((size_t)32 * 1024 * 4);
    int*   fpsidx = (int*)alloc((size_t)32 * 512 * 4);
    int*   gi     = (int*)alloc((size_t)32 * 512 * 128 * 4);
    float* stats  = (float*)alloc(2 * 1024 * 4);
    float* hn     = (float*)alloc((size_t)32 * 2048 * 4);
    float* tnv    = (float*)alloc(32 * 4);
    float* simf   = (float*)alloc((size_t)32 * 2048 * 32 * 4);

    struct Lyr { const _Float16* WT; const float* b; const float* g; const float* beta;
                 int K, N, Kpad, Npad; };
    auto ceil32 = [](int x) { return (x + 31) & ~31; };
    auto prep = [&](int widx, int K, int N) -> Lyr {
        Lyr L; L.K = K; L.N = N; L.Kpad = ceil32(K); L.Npad = ceil32(N);
        _Float16* wt = (_Float16*)alloc((size_t)L.Kpad * L.Npad * 2);
        float* bb = (float*)alloc(L.Npad * 4);
        float* gg = (float*)alloc(L.Npad * 4);
        float* be = (float*)alloc(L.Npad * 4);
        int tot = L.Kpad * L.Npad;
        prep_w_kernel<<<(tot + 255) / 256, 256, 0, stream>>>(P(widx), K, N, L.Kpad, L.Npad, wt);
        prep_vec_kernel<<<(L.Npad + 63) / 64, 64, 0, stream>>>(P(widx + 1), N, L.Npad, bb);
        prep_vec_kernel<<<(L.Npad + 63) / 64, 64, 0, stream>>>(P(widx + 2), N, L.Npad, gg);
        prep_vec_kernel<<<(L.Npad + 63) / 64, 64, 0, stream>>>(P(widx + 3), N, L.Npad, be);
        L.WT = wt; L.b = bb; L.g = gg; L.beta = be;
        return L;
    };

    // fused GEMM + BN(+ReLU); M must be a multiple of 32
    auto gemm_bn = [&](const _Float16* X, _Float16* Y, int M, const Lyr& L, bool relu) {
        hipMemsetAsync(stats, 0, (size_t)2 * L.Npad * 4, stream);
        dim3 g(M / 32, (L.Npad + 63) / 64);
        gemm_wmma_kernel<<<g, 256, 0, stream>>>(X, L.WT, L.b, Y, nullptr, stats, M, L.Kpad, L.Npad);
        size_t tot8 = (size_t)M * L.Npad / 8;
        bn_act_kernel<<<(tot8 + 255) / 256, 256, 0, stream>>>(Y, stats, L.g, L.beta,
                                                              1.0f / M, L.Npad, tot8, relu ? 1 : 0);
    };

    // ---- parameter prep (insertion-order flattening of the params dict) ----
    Lyr s1[3][3] = {
        { prep(2, 6, 32),   prep(6, 32, 32),   prep(10, 32, 64) },
        { prep(14, 6, 64),  prep(18, 64, 64),  prep(22, 64, 128) },
        { prep(26, 6, 64),  prep(30, 64, 96),  prep(34, 96, 128) } };
    Lyr s2[2][3] = {
        { prep(38, 323, 128), prep(42, 128, 128), prep(46, 128, 256) },
        { prep(50, 323, 128), prep(54, 128, 196), prep(58, 196, 256) } };
    Lyr s3[3]  = { prep(62, 515, 256), prep(66, 256, 512), prep(70, 512, 1024) };
    Lyr f3[2]  = { prep(74, 1536, 256), prep(78, 256, 256) };
    Lyr f2[2]  = { prep(82, 576, 256),  prep(86, 256, 128) };
    Lyr f1[2]  = { prep(90, 134, 128),  prep(94, 128, 128) };
    Lyr headL  = prep(98, 128, 512);
    const float* logit_scale = P(102);

    _Float16* textWT = (_Float16*)alloc((size_t)32 * 512 * 2);
    prep_w_kernel<<<(32 * 512 + 255) / 256, 256, 0, stream>>>(text, 512, 19, 512, 32, textWT);

    // ---- pts = transpose(xyz) ----
    transpose_xyz_kernel<<<(32 * 2048 * 3 + 255) / 256, 256, 0, stream>>>(xyz, pts);

    // ---- SA1 (npoint=512, radii .1/.2/.4, nsample 32/64/128) ----
    fps_kernel<<<32, 256, 0, stream>>>(pts, 2048, 512, fpsidx);
    gather_xyz_kernel<<<(32 * 512 + 255) / 256, 256, 0, stream>>>(pts, fpsidx, 2048, 512, l1x, 32 * 512);
    {
        const float rad[3] = { 0.1f, 0.2f, 0.4f };
        const int   ns[3]  = { 32, 64, 128 };
        int coff = 0;
        for (int br = 0; br < 3; br++) {
            int K = ns[br];
            size_t rows = (size_t)32 * 512 * K;
            ball_query_kernel<<<(32 * 512 + 255) / 256, 256, 0, stream>>>(
                pts, l1x, 2048, 512, K, rad[br] * rad[br], gi, 32 * 512);
            group_gather_kernel<<<(rows + 255) / 256, 256, 0, stream>>>(
                pts, 2048, 3, pts, l1x, gi, 512, K, s1[br][0].Kpad, ACT0, rows);
            _Float16* curb = ACT0; _Float16* nxt = ACT1;
            for (int l = 0; l < 3; l++) {
                gemm_bn(curb, nxt, (int)rows, s1[br][l], true);
                _Float16* t = curb; curb = nxt; nxt = t;
            }
            size_t ptot = (size_t)32 * 512 * s1[br][2].N;
            maxpool_kernel<<<(ptot + 255) / 256, 256, 0, stream>>>(
                curb, K, s1[br][2].N, s1[br][2].Npad, l1f, 320, coff, (size_t)32 * 512);
            coff += s1[br][2].N;
        }
    }

    // ---- SA2 (npoint=128, radii .4/.8, nsample 64/128) ----
    fps_kernel<<<32, 256, 0, stream>>>(l1x, 512, 128, fpsidx);
    gather_xyz_kernel<<<(32 * 128 + 255) / 256, 256, 0, stream>>>(l1x, fpsidx, 512, 128, l2x, 32 * 128);
    {
        const float rad[2] = { 0.4f, 0.8f };
        const int   ns[2]  = { 64, 128 };
        int coff = 0;
        for (int br = 0; br < 2; br++) {
            int K = ns[br];
            size_t rows = (size_t)32 * 128 * K;
            ball_query_kernel<<<(32 * 128 + 255) / 256, 256, 0, stream>>>(
                l1x, l2x, 512, 128, K, rad[br] * rad[br], gi, 32 * 128);
            group_gather_kernel<<<(rows + 255) / 256, 256, 0, stream>>>(
                l1f, 512, 320, l1x, l2x, gi, 128, K, s2[br][0].Kpad, ACT0, rows);
            _Float16* curb = ACT0; _Float16* nxt = ACT1;
            for (int l = 0; l < 3; l++) {
                gemm_bn(curb, nxt, (int)rows, s2[br][l], true);
                _Float16* t = curb; curb = nxt; nxt = t;
            }
            size_t ptot = (size_t)32 * 128 * s2[br][2].N;
            maxpool_kernel<<<(ptot + 255) / 256, 256, 0, stream>>>(
                curb, K, s2[br][2].N, s2[br][2].Npad, l2f, 512, coff, (size_t)32 * 128);
            coff += s2[br][2].N;
        }
    }

    // ---- SA3 group-all ----
    sa3_build_kernel<<<(4096 + 255) / 256, 256, 0, stream>>>(l2x, l2f, ACT0);
    {
        _Float16* curb = ACT0; _Float16* nxt = ACT1;
        for (int l = 0; l < 3; l++) {
            gemm_bn(curb, nxt, 4096, s3[l], true);
            _Float16* t = curb; curb = nxt; nxt = t;
        }
        maxpool_kernel<<<((size_t)32 * 1024 + 255) / 256, 256, 0, stream>>>(
            curb, 128, 1024, 1024, l3f, 1024, 0, 32);
    }

    // ---- FP3: concat(l2f, broadcast l3f) -> MLP [1536,256,256] ----
    fp3_build_kernel<<<(4096 + 255) / 256, 256, 0, stream>>>(l2f, l3f, ACT0);
    gemm_bn(ACT0, ACT1, 4096, f3[0], true);
    gemm_bn(ACT1, ACT0, 4096, f3[1], true);   // fp3 out: ACT0 [4096,256] f16

    // ---- FP2: 3-NN(512<-128) interp + concat l1f -> MLP [576,256,128] ----
    copy_cols_kernel<<<((size_t)32 * 512 + 255) / 256, 256, 0, stream>>>(
        l1f, 320, 320, ACT1, 576, (size_t)32 * 512);
    knn3_interp_kernel<<<(32 * 512 + 255) / 256, 256, 0, stream>>>(
        l1x, 512, l2x, 128, ACT0, 256, 256, ACT1, 576, 320, 32 * 512);
    gemm_bn(ACT1, ACT0, 16384, f2[0], true);
    gemm_bn(ACT0, ACT1, 16384, f2[1], true);  // fp2 out: ACT1 [16384,128] f16

    // ---- FP1: 3-NN(2048<-512) interp + concat(pts,pts) -> MLP [134,128,128] ----
    fp1_build_kernel<<<(32 * 2048 + 255) / 256, 256, 0, stream>>>(pts, ACT0);
    knn3_interp_kernel<<<(32 * 2048 + 255) / 256, 256, 0, stream>>>(
        pts, 2048, l1x, 512, ACT1, 128, 128, ACT0, 160, 6, 32 * 2048);
    gemm_bn(ACT0, ACT1, 65536, f1[0], true);
    gemm_bn(ACT1, ACT0, 65536, f1[1], true);  // l0f: ACT0 [65536,128] f16

    // ---- head (linear + BN, no ReLU) ----
    gemm_bn(ACT0, ACT1, 65536, headL, false); // h: ACT1 [65536,512] f16

    // ---- cosine similarity logits + log_softmax ----
    rownorm_kernel<<<(65536 + 255) / 256, 256, 0, stream>>>(ACT1, 512, hn, 65536);
    tn_kernel<<<1, 32, 0, stream>>>(text, tnv);
    {
        dim3 g(65536 / 32, 1);
        gemm_wmma_kernel<<<g, 256, 0, stream>>>(ACT1, textWT, nullptr, nullptr, simf,
                                                nullptr, 65536, 512, 32);
    }
    logits_kernel<<<(32 * 2048 + 255) / 256, 256, 0, stream>>>(
        simf, hn, tnv, logit_scale, (float*)d_out);
}
